// StatesBottleneck_1924145349109
// MI455X (gfx1250) — matrix operand order repaired and checked by reading, not compiled, not run on hardware
//
#include <hip/hip_runtime.h>
#include <math.h>

typedef __attribute__((ext_vector_type(16))) _Float16 v16h;
typedef __attribute__((ext_vector_type(8)))  float    v8f;

#define N_NODES    100000
#define N_EDGES    1600000
#define HDIM       128
#define NUM_GRAPHS 64

// Monotonic unsigned encoding of float (order-preserving), for atomic max.
#define ENC_NEG_INF 0x007FFFFFu
__device__ __forceinline__ unsigned encOrd(float f) {
  unsigned u = __float_as_uint(f);
  return (u & 0x80000000u) ? ~u : (u | 0x80000000u);
}
__device__ __forceinline__ float decOrd(unsigned e) {
  return __uint_as_float((e & 0x80000000u) ? (e ^ 0x80000000u) : ~e);
}

__device__ __forceinline__ float blockReduceSum(float v, float* red) {
  int t = threadIdx.x;
  red[t] = v;
  __syncthreads();
  for (int s = blockDim.x >> 1; s > 0; s >>= 1) {
    if (t < s) red[t] += red[t + s];
    __syncthreads();
  }
  float r = red[0];
  __syncthreads();
  return r;
}

// ---------------------------------------------------------------------------
// 0. Init accumulators
// ---------------------------------------------------------------------------
__global__ void init_kernel(unsigned* enc, float* sums, float* scalars) {
  int t = threadIdx.x;
  if (t < 192) enc[t] = ENC_NEG_INF;   // 3 arrays of 64 encoded segment maxes
  if (t < 192) sums[t] = 0.f;          // 3 arrays of 64 exp-sums
  if (t < 8)   scalars[t] = 0.f;       // loss accumulators + count
}

// ---------------------------------------------------------------------------
// 1. Node GEMM: logits = node_fts @ W_node.T + b  (WMMA f16->f32)
//    Fused: BCE partial for col 1, segment-max for cols 0 and 2.
// ---------------------------------------------------------------------------
__global__ __launch_bounds__(256)
void node_gemm_kernel(const float* __restrict__ fts,
                      const float* __restrict__ W,
                      const float* __restrict__ bias,
                      const float* __restrict__ hints,
                      const int*   __restrict__ batch,
                      float*       __restrict__ logits_out,
                      unsigned*    __restrict__ enc_max,   // [2*64]: col0, col2
                      float*       __restrict__ bce_accum) {
  __shared__ unsigned ldsMax[2 * NUM_GRAPHS];
  __shared__ float red[256];
  for (int i = threadIdx.x; i < 2 * NUM_GRAPHS; i += blockDim.x) ldsMax[i] = ENC_NEG_INF;
  __syncthreads();

  const int lane = threadIdx.x & 31;
  const int wave = threadIdx.x >> 5;
  const int tile = blockIdx.x * (blockDim.x >> 5) + wave;  // wave-uniform
  const int half = lane >> 4;
  const int lm   = lane & 15;
  float bcePartial = 0.f;

  if (tile * 16 < N_NODES) {
    const long long row0 = (long long)tile * 16;
    const float* arow = fts + (row0 + lm) * HDIM;  // M = lm for all lanes
    v8f c = {};
#pragma unroll
    for (int k0 = 0; k0 < HDIM; k0 += 32) {
      // A (16x32 f16): h[0..7]=K=k0+half*8+0..7, h[8..15]=K=k0+16+half*8+0..7
      v16h a;
      {
        const float4* p1 = (const float4*)(arow + k0 + half * 8);
        const float4* p2 = (const float4*)(arow + k0 + 16 + half * 8);
        float4 f0 = p1[0], f1 = p1[1], f2 = p2[0], f3 = p2[1];
        a[0]=(_Float16)f0.x;  a[1]=(_Float16)f0.y;  a[2]=(_Float16)f0.z;  a[3]=(_Float16)f0.w;
        a[4]=(_Float16)f1.x;  a[5]=(_Float16)f1.y;  a[6]=(_Float16)f1.z;  a[7]=(_Float16)f1.w;
        a[8]=(_Float16)f2.x;  a[9]=(_Float16)f2.y;  a[10]=(_Float16)f2.z; a[11]=(_Float16)f2.w;
        a[12]=(_Float16)f3.x; a[13]=(_Float16)f3.y; a[14]=(_Float16)f3.z; a[15]=(_Float16)f3.w;
      }
      // B (32x16 f16): N = lm, h[0..7]=K=k0+half*8+0..7, h[8..15]=K=k0+16+half*8+0..7; zero for N >= 3
      v16h b = {};
      if (lm < 3) {
        const float4* wp1 = (const float4*)(W + lm * HDIM + k0 + half * 8);
        const float4* wp2 = (const float4*)(W + lm * HDIM + k0 + 16 + half * 8);
        float4 w0 = wp1[0], w1 = wp1[1], w2 = wp2[0], w3 = wp2[1];
        b[0]=(_Float16)w0.x;  b[1]=(_Float16)w0.y;  b[2]=(_Float16)w0.z;  b[3]=(_Float16)w0.w;
        b[4]=(_Float16)w1.x;  b[5]=(_Float16)w1.y;  b[6]=(_Float16)w1.z;  b[7]=(_Float16)w1.w;
        b[8]=(_Float16)w2.x;  b[9]=(_Float16)w2.y;  b[10]=(_Float16)w2.z; b[11]=(_Float16)w2.w;
        b[12]=(_Float16)w3.x; b[13]=(_Float16)w3.y; b[14]=(_Float16)w3.z; b[15]=(_Float16)w3.w;
      }
      c = __builtin_amdgcn_wmma_f32_16x16x32_f16(false, a, false, b, (short)0, c, false, false);
    }
    if (lm < 3) {
      float bv = bias[lm];
#pragma unroll
      for (int r = 0; r < 8; ++r) {
        long long row = row0 + half * 8 + r;   // C: VGPR r -> M = r + half*8
        float l = c[r] + bv;
        logits_out[row * 3 + lm] = l;
        if (lm == 1) {  // MASK column -> BCE
          float gt = hints[row * 3 + 1];
          bcePartial += fmaxf(l, 0.f) - l * gt + log1pf(expf(-fabsf(l)));
        } else {        // POINTER columns -> segment max
          int seg = batch[row];
          int arr = (lm == 0) ? 0 : 1;
          atomicMax(&ldsMax[arr * NUM_GRAPHS + seg], encOrd(l));
        }
      }
    }
  }
  __syncthreads();
  for (int i = threadIdx.x; i < 2 * NUM_GRAPHS; i += blockDim.x) {
    unsigned v = ldsMax[i];
    if (v != ENC_NEG_INF) atomicMax(&enc_max[i], v);
  }
  float tot = blockReduceSum(bcePartial, red);
  if (threadIdx.x == 0 && tot != 0.f) atomicAdd(bce_accum, tot);
}

// ---------------------------------------------------------------------------
// 2. Edge GEMM: logits = edge_fts @ W_edge.T + b.
//    Fused: BCE partial for col 0, segment-max (batch[edge_src]) for col 1.
// ---------------------------------------------------------------------------
__global__ __launch_bounds__(256)
void edge_gemm_kernel(const float* __restrict__ fts,
                      const float* __restrict__ W,
                      const float* __restrict__ bias,
                      const float* __restrict__ hints,
                      const int*   __restrict__ batch,
                      const int*   __restrict__ edge_src,
                      float*       __restrict__ logits_out,
                      unsigned*    __restrict__ enc_max,   // [64]
                      float*       __restrict__ bce_accum) {
  __shared__ unsigned ldsMax[NUM_GRAPHS];
  __shared__ float red[256];
  for (int i = threadIdx.x; i < NUM_GRAPHS; i += blockDim.x) ldsMax[i] = ENC_NEG_INF;
  __syncthreads();

  const int lane = threadIdx.x & 31;
  const int wave = threadIdx.x >> 5;
  const int tile = blockIdx.x * (blockDim.x >> 5) + wave;
  const int half = lane >> 4;
  const int lm   = lane & 15;
  float bcePartial = 0.f;

  if (tile * 16 < N_EDGES) {
    const long long row0 = (long long)tile * 16;
    const float* arow = fts + (row0 + lm) * HDIM;
    v8f c = {};
#pragma unroll
    for (int k0 = 0; k0 < HDIM; k0 += 32) {
      v16h a;
      {
        const float4* p1 = (const float4*)(arow + k0 + half * 8);
        const float4* p2 = (const float4*)(arow + k0 + 16 + half * 8);
        float4 f0 = p1[0], f1 = p1[1], f2 = p2[0], f3 = p2[1];
        a[0]=(_Float16)f0.x;  a[1]=(_Float16)f0.y;  a[2]=(_Float16)f0.z;  a[3]=(_Float16)f0.w;
        a[4]=(_Float16)f1.x;  a[5]=(_Float16)f1.y;  a[6]=(_Float16)f1.z;  a[7]=(_Float16)f1.w;
        a[8]=(_Float16)f2.x;  a[9]=(_Float16)f2.y;  a[10]=(_Float16)f2.z; a[11]=(_Float16)f2.w;
        a[12]=(_Float16)f3.x; a[13]=(_Float16)f3.y; a[14]=(_Float16)f3.z; a[15]=(_Float16)f3.w;
      }
      v16h b = {};
      if (lm < 2) {
        const float4* wp1 = (const float4*)(W + lm * HDIM + k0 + half * 8);
        const float4* wp2 = (const float4*)(W + lm * HDIM + k0 + 16 + half * 8);
        float4 w0 = wp1[0], w1 = wp1[1], w2 = wp2[0], w3 = wp2[1];
        b[0]=(_Float16)w0.x;  b[1]=(_Float16)w0.y;  b[2]=(_Float16)w0.z;  b[3]=(_Float16)w0.w;
        b[4]=(_Float16)w1.x;  b[5]=(_Float16)w1.y;  b[6]=(_Float16)w1.z;  b[7]=(_Float16)w1.w;
        b[8]=(_Float16)w2.x;  b[9]=(_Float16)w2.y;  b[10]=(_Float16)w2.z; b[11]=(_Float16)w2.w;
        b[12]=(_Float16)w3.x; b[13]=(_Float16)w3.y; b[14]=(_Float16)w3.z; b[15]=(_Float16)w3.w;
      }
      c = __builtin_amdgcn_wmma_f32_16x16x32_f16(false, a, false, b, (short)0, c, false, false);
    }
    if (lm < 2) {
      float bv = bias[lm];
#pragma unroll
      for (int r = 0; r < 8; ++r) {
        long long row = row0 + half * 8 + r;
        float l = c[r] + bv;
        logits_out[row * 2 + lm] = l;
        if (lm == 0) {  // MASK -> BCE
          float gt = hints[row * 2 + 0];
          bcePartial += fmaxf(l, 0.f) - l * gt + log1pf(expf(-fabsf(l)));
        } else {        // EDGE_MASK_ONE -> segment max over batch[edge_src]
          int seg = batch[edge_src[row]];
          atomicMax(&ldsMax[seg], encOrd(l));
        }
      }
    }
  }
  __syncthreads();
  for (int i = threadIdx.x; i < NUM_GRAPHS; i += blockDim.x) {
    unsigned v = ldsMax[i];
    if (v != ENC_NEG_INF) atomicMax(&enc_max[i], v);
  }
  float tot = blockReduceSum(bcePartial, red);
  if (threadIdx.x == 0 && tot != 0.f) atomicAdd(bce_accum, tot);
}

// ---------------------------------------------------------------------------
// 3. Count (batch == 0) -> EDGE_MASK_ONE weight
// ---------------------------------------------------------------------------
__global__ __launch_bounds__(256)
void count_kernel(const int* __restrict__ batch, float* __restrict__ scalars) {
  __shared__ float red[256];
  float c = 0.f;
  for (int i = blockIdx.x * blockDim.x + threadIdx.x; i < N_NODES; i += gridDim.x * blockDim.x)
    c += (batch[i] == 0) ? 1.f : 0.f;
  float t = blockReduceSum(c, red);
  if (threadIdx.x == 0 && t != 0.f) atomicAdd(&scalars[5], t);
}

// ---------------------------------------------------------------------------
// 4. Pass 2: segment exp-sums
// ---------------------------------------------------------------------------
__global__ __launch_bounds__(256)
void node_sum_kernel(const float* __restrict__ logits,
                     const int*   __restrict__ batch,
                     const unsigned* __restrict__ enc_max,  // [2*64]
                     float* __restrict__ sums) {            // [2*64]
  __shared__ float ldsSum[2 * NUM_GRAPHS];
  for (int i = threadIdx.x; i < 2 * NUM_GRAPHS; i += blockDim.x) ldsSum[i] = 0.f;
  __syncthreads();
  for (int i = blockIdx.x * blockDim.x + threadIdx.x; i < N_NODES; i += gridDim.x * blockDim.x) {
    int seg = batch[i];
    unsigned e0 = enc_max[seg], e2 = enc_max[NUM_GRAPHS + seg];
    float m0 = (e0 == ENC_NEG_INF) ? 0.f : decOrd(e0);
    float m2 = (e2 == ENC_NEG_INF) ? 0.f : decOrd(e2);
    atomicAdd(&ldsSum[seg],              expf(logits[i * 3 + 0] - m0));
    atomicAdd(&ldsSum[NUM_GRAPHS + seg], expf(logits[i * 3 + 2] - m2));
  }
  __syncthreads();
  for (int i = threadIdx.x; i < 2 * NUM_GRAPHS; i += blockDim.x) {
    float v = ldsSum[i];
    if (v != 0.f) atomicAdd(&sums[i], v);
  }
}

__global__ __launch_bounds__(256)
void edge_sum_kernel(const float* __restrict__ logits,
                     const int*   __restrict__ batch,
                     const int*   __restrict__ edge_src,
                     const unsigned* __restrict__ enc_max,  // [64]
                     float* __restrict__ sums) {            // [64]
  __shared__ float ldsSum[NUM_GRAPHS];
  for (int i = threadIdx.x; i < NUM_GRAPHS; i += blockDim.x) ldsSum[i] = 0.f;
  __syncthreads();
  for (int i = blockIdx.x * blockDim.x + threadIdx.x; i < N_EDGES; i += gridDim.x * blockDim.x) {
    int seg = batch[edge_src[i]];
    unsigned e1 = enc_max[seg];
    float m1 = (e1 == ENC_NEG_INF) ? 0.f : decOrd(e1);
    atomicAdd(&ldsSum[seg], expf(logits[i * 2 + 1] - m1));
  }
  __syncthreads();
  for (int i = threadIdx.x; i < NUM_GRAPHS; i += blockDim.x) {
    float v = ldsSum[i];
    if (v != 0.f) atomicAdd(&sums[i], v);
  }
}

// ---------------------------------------------------------------------------
// 5. Pass 3: predictions + pointer-loss partials
// ---------------------------------------------------------------------------
__global__ __launch_bounds__(256)
void node_final_kernel(const float* __restrict__ logits,
                       const float* __restrict__ hints,
                       const int*   __restrict__ batch,
                       const unsigned* __restrict__ enc_max,  // [2*64]
                       const float* __restrict__ sums,        // [2*64]
                       float* __restrict__ out,               // [N_NODES*3]
                       float* __restrict__ scalars) {
  __shared__ float red[256];
  float p0 = 0.f, p2 = 0.f;
  for (int i = blockIdx.x * blockDim.x + threadIdx.x; i < N_NODES; i += gridDim.x * blockDim.x) {
    int seg = batch[i];
    unsigned e0 = enc_max[seg], e2 = enc_max[NUM_GRAPHS + seg];
    float rm0 = decOrd(e0), rm2 = decOrd(e2);
    float m0 = (e0 == ENC_NEG_INF) ? 0.f : rm0;
    float m2 = (e2 == ENC_NEG_INF) ? 0.f : rm2;
    float lse0 = m0 + logf(sums[seg]);
    float lse2 = m2 + logf(sums[NUM_GRAPHS + seg]);
    float l0 = logits[i * 3 + 0], l1 = logits[i * 3 + 1], l2 = logits[i * 3 + 2];
    out[i * 3 + 0] = (l0 >= rm0) ? 1.f : 0.f;
    out[i * 3 + 1] = (l1 > 0.f) ? 1.f : 0.f;
    out[i * 3 + 2] = (l2 >= rm2) ? 1.f : 0.f;
    p0 -= hints[i * 3 + 0] * (l0 - lse0);
    p2 -= hints[i * 3 + 2] * (l2 - lse2);
  }
  float t0 = blockReduceSum(p0, red);
  float t2 = blockReduceSum(p2, red);
  if (threadIdx.x == 0) {
    if (t0 != 0.f) atomicAdd(&scalars[0], t0);
    if (t2 != 0.f) atomicAdd(&scalars[2], t2);
  }
}

__global__ __launch_bounds__(256)
void edge_final_kernel(const float* __restrict__ logits,
                       const float* __restrict__ hints,
                       const int*   __restrict__ batch,
                       const int*   __restrict__ edge_src,
                       const unsigned* __restrict__ enc_max,  // [64]
                       const float* __restrict__ sums,        // [64]
                       float* __restrict__ out,               // [N_EDGES*2]
                       float* __restrict__ scalars) {
  __shared__ float red[256];
  float pe = 0.f;
  for (int i = blockIdx.x * blockDim.x + threadIdx.x; i < N_EDGES; i += gridDim.x * blockDim.x) {
    int seg = batch[edge_src[i]];
    unsigned e1 = enc_max[seg];
    float rm1 = decOrd(e1);
    float m1 = (e1 == ENC_NEG_INF) ? 0.f : rm1;
    float lse1 = m1 + logf(sums[seg]);
    float l0 = logits[i * 2 + 0], l1 = logits[i * 2 + 1];
    out[i * 2 + 0] = (l0 > 0.f) ? 1.f : 0.f;
    out[i * 2 + 1] = (l1 >= rm1) ? 1.f : 0.f;
    pe -= hints[i * 2 + 1] * (l1 - lse1);
  }
  float te = blockReduceSum(pe, red);
  if (threadIdx.x == 0 && te != 0.f) atomicAdd(&scalars[4], te);
}

// ---------------------------------------------------------------------------
// 6. Final scalar loss
// ---------------------------------------------------------------------------
__global__ void finalize_kernel(const float* __restrict__ scalars, float* __restrict__ out_loss) {
  if (threadIdx.x == 0) {
    float loss = scalars[0] * (1.f / NUM_GRAPHS)        // node ptr col0 (w=1)
               + scalars[1] * (1.f / N_NODES)           // node BCE col1 (mean)
               + scalars[2] * (1.f / NUM_GRAPHS)        // node ptr col2 (w=1)
               + scalars[3] * (1.f / N_EDGES)           // edge BCE col0 (mean)
               + scalars[5] * scalars[4] * (1.f / NUM_GRAPHS);  // edge ptr, w=count(batch==0)
    *out_loss = loss;
  }
}

// ---------------------------------------------------------------------------
extern "C" void kernel_launch(void* const* d_in, const int* in_sizes, int n_in,
                              void* d_out, int out_size, void* d_ws, size_t ws_size,
                              hipStream_t stream) {
  const float* node_fts   = (const float*)d_in[0];
  const float* edge_fts   = (const float*)d_in[1];
  const float* W_node     = (const float*)d_in[2];
  const float* b_node     = (const float*)d_in[3];
  const float* W_edge     = (const float*)d_in[4];
  const float* b_edge     = (const float*)d_in[5];
  const float* node_hints = (const float*)d_in[6];
  const float* edge_hints = (const float*)d_in[7];
  const int*   batch      = (const int*)d_in[8];
  const int*   edge_src   = (const int*)d_in[9];
  float* out = (float*)d_out;

  // Workspace: node_logits | edge_logits | 3x64 enc-max | 3x64 sums | 8 scalars
  float*    ws          = (float*)d_ws;
  float*    node_logits = ws;
  float*    edge_logits = ws + (size_t)N_NODES * 3;
  unsigned* enc         = (unsigned*)(ws + (size_t)N_NODES * 3 + (size_t)N_EDGES * 2);
  float*    sums        = (float*)(enc + 192);
  float*    scalars     = sums + 192;

  init_kernel<<<1, 256, 0, stream>>>(enc, sums, scalars);

  int nodeBlocks = (N_NODES / 16 + 7) / 8;   // 8 waves (16-row tiles) per 256-thread block
  node_gemm_kernel<<<nodeBlocks, 256, 0, stream>>>(node_fts, W_node, b_node, node_hints,
                                                   batch, node_logits, enc, &scalars[1]);
  int edgeBlocks = (N_EDGES / 16 + 7) / 8;
  edge_gemm_kernel<<<edgeBlocks, 256, 0, stream>>>(edge_fts, W_edge, b_edge, edge_hints,
                                                   batch, edge_src, edge_logits, enc + 128, &scalars[3]);
  count_kernel<<<256, 256, 0, stream>>>(batch, scalars);
  node_sum_kernel<<<512, 256, 0, stream>>>(node_logits, batch, enc, sums);
  edge_sum_kernel<<<2048, 256, 0, stream>>>(edge_logits, batch, edge_src, enc + 128, sums + 128);
  node_final_kernel<<<512, 256, 0, stream>>>(node_logits, node_hints, batch, enc, sums, out, scalars);
  edge_final_kernel<<<2048, 256, 0, stream>>>(edge_logits, edge_hints, batch, edge_src,
                                              enc + 128, sums + 128, out + (size_t)N_NODES * 3, scalars);
  finalize_kernel<<<1, 1, 0, stream>>>(scalars, out + (size_t)out_size - 1);
}
